// GptOssAttention_36661840839148
// MI455X (gfx1250) — matrix operand order repaired and checked by reading, not compiled, not run on hardware
//
#include <hip/hip_runtime.h>
#include <hip/hip_bf16.h>
#include <math.h>

// ---------------------------------------------------------------------------
// Types for CDNA5 WMMA (gfx1250, wave32)
// ---------------------------------------------------------------------------
typedef __attribute__((ext_vector_type(16))) __bf16 v16bf;
typedef __attribute__((ext_vector_type(8)))  float  v8f;

union Frag {
    uint4 q[2];   // 32 bytes = 16 bf16
    v16bf v;
};

// CDNA5 async global->LDS copy path (ASYNCcnt), if the toolchain exposes it.
#if defined(__has_builtin)
#if __has_builtin(__builtin_amdgcn_global_load_async_to_lds_b128) && \
    __has_builtin(__builtin_amdgcn_s_wait_asynccnt)
#define HAVE_ASYNC_LDS 1
#endif
#endif
#ifndef HAVE_ASYNC_LDS
#define HAVE_ASYNC_LDS 0
#endif

#if HAVE_ASYNC_LDS
// Builtin prototype (from clang diagnostics): (int4 AS1*, int4 AS3*, Imm, Imm)
typedef int async_v4i __attribute__((ext_vector_type(4)));
typedef __attribute__((address_space(1))) async_v4i* async_gptr;
typedef __attribute__((address_space(3))) async_v4i* async_lptr;
#endif

__device__ __forceinline__ unsigned short f2bf(float f) {
    union { float f; unsigned u; } v; v.f = f;
    unsigned r = v.u + 0x7FFFu + ((v.u >> 16) & 1u);   // round-to-nearest-even
    return (unsigned short)(r >> 16);
}
__device__ __forceinline__ float bf2f(unsigned short u) {
    union { unsigned u; float f; } v; v.u = ((unsigned)u) << 16;
    return v.f;
}
__device__ __forceinline__ unsigned pack2bf(float lo, float hi) {
    return (unsigned)f2bf(lo) | ((unsigned)f2bf(hi) << 16);
}

__device__ __forceinline__ v8f wmma_bf16(v16bf a, v16bf b, v8f c) {
    // D = A(16x32 bf16) * B(32x16 bf16) + C(16x16 f32)
    return __builtin_amdgcn_wmma_f32_16x16x32_bf16(
        /*neg_a=*/false, a, /*neg_b=*/false, b,
        /*c_mod=*/(short)0, c, /*reuse_a=*/false, /*reuse_b=*/false);
}

// ---------------------------------------------------------------------------
// Problem constants
// ---------------------------------------------------------------------------
#define HID_DIM 2880
#define NQ_H    64
#define NKV_H   8
#define S_TOK   8192

// Workspace layout (bytes, all 16B-aligned)
static const size_t OFF_HID = 0;                                       // 8192*2880 bf16
static const size_t OFF_WQ  = OFF_HID + (size_t)S_TOK * HID_DIM * 2;   // 2880*4096 bf16
static const size_t OFF_WK  = OFF_WQ  + (size_t)HID_DIM * 4096 * 2;    // 2880*512 bf16
static const size_t OFF_WV  = OFF_WK  + (size_t)HID_DIM * 512 * 2;
static const size_t OFF_WO  = OFF_WV  + (size_t)HID_DIM * 512 * 2;     // 4096*2880 bf16
static const size_t OFF_Q   = OFF_WO  + (size_t)4096 * HID_DIM * 2;    // 8192*4096 bf16
static const size_t OFF_K   = OFF_Q   + (size_t)S_TOK * 4096 * 2;      // 8192*512 bf16
static const size_t OFF_V   = OFF_K   + (size_t)S_TOK * 512 * 2;
static const size_t OFF_ATT = OFF_V   + (size_t)S_TOK * 512 * 2;       // 8192*4096 bf16

// ---------------------------------------------------------------------------
// fp32 -> bf16 conversion, 8 elements per thread (n must be divisible by 8)
// ---------------------------------------------------------------------------
__global__ __launch_bounds__(256)
void f32_to_bf16_kernel(const float* __restrict__ src,
                        unsigned short* __restrict__ dst, long long n8) {
    long long i = (long long)blockIdx.x * blockDim.x + threadIdx.x;
    if (i >= n8) return;
    const float4 a = *(const float4*)(src + i * 8);
    const float4 b = *(const float4*)(src + i * 8 + 4);
    uint4 o;
    o.x = pack2bf(a.x, a.y);
    o.y = pack2bf(a.z, a.w);
    o.z = pack2bf(b.x, b.y);
    o.w = pack2bf(b.z, b.w);
    *(uint4*)(dst + i * 8) = o;
}

// ---------------------------------------------------------------------------
// bf16 WMMA GEMM: C[M,N] = A[M,K] * B[K,N] + bias[N]
// Compile-time N, K, STORE_BF16 so all strides fold into immediates.
// Block = 256 threads (8 waves). Macro-tile 128(M) x 64(N), BK = 32.
// Wave tile 32x32 (2x2 WMMA C-tiles). Waves arranged 4(M) x 2(N).
// A tile: async global->LDS copy (ASYNCcnt) when available.
// ---------------------------------------------------------------------------
template <int N, int K, bool STORE_BF16>
__global__ __launch_bounds__(256)
void gemm_bf16_kernel(const unsigned short* __restrict__ A,
                      const unsigned short* __restrict__ B,
                      const float* __restrict__ bias,
                      void* __restrict__ Cout)
{
    __shared__ __align__(16) unsigned short As[128 * 32];   // [m][k] row-major
    __shared__ __align__(16) unsigned short BsT[64 * 32];   // [n][k] (B transposed)

    const int t    = threadIdx.x;
    const int lane = t & 31;
    const int wave = t >> 5;
    const int m0   = blockIdx.y * 128;
    const int n0   = blockIdx.x * 64;
    const int wm   = (wave >> 1) * 32;
    const int wn   = (wave & 1) * 32;
    const int l15  = lane & 15;
    const int hi   = lane >> 4;     // 0 / 1
    const int kbA  = hi * 16;       // A-frag: K run  [kbA,kbA+16) contiguous
    const int ksB  = hi * 16;       // B-frag: K run  [ksB,ksB+16) contiguous

    // Global->LDS load roles
    const int arow = t >> 1;        // 0..127
    const int acol = (t & 1) * 16;  // 0 / 16
    const int bn   = t & 63;        // 0..63
    const int bk8  = (t >> 6) * 8;  // 0,8,16,24

    const v8f vzero = {0.f,0.f,0.f,0.f,0.f,0.f,0.f,0.f};
    v8f acc[2][2];
    #pragma unroll
    for (int i = 0; i < 2; ++i)
        #pragma unroll
        for (int j = 0; j < 2; ++j) acc[i][j] = vzero;

    const unsigned short* aPtr = A + (size_t)(m0 + arow) * K + acol;
    const unsigned short* bPtr = B + (size_t)bk8 * N + (n0 + bn);
    unsigned short* asDst = As + arow * 32 + acol;   // mirrors aPtr byte-for-byte

    for (int k0 = 0; k0 < K; k0 += 32) {
#if !HAVE_ASYNC_LDS
        uint4 a0 = *(const uint4*)(aPtr);
        uint4 a1 = *(const uint4*)(aPtr + 8);
#endif
        unsigned short bv[8];
        #pragma unroll
        for (int kk = 0; kk < 8; ++kk)
            bv[kk] = bPtr[(size_t)kk * N];
        if (k0 + 32 < K)
            __builtin_prefetch(aPtr + 32, 0, 1);   // global_prefetch_b8

        __syncthreads();   // previous iteration's LDS reads done
#if HAVE_ASYNC_LDS
        // Direct global->LDS DMA of this thread's 32 bytes of the A tile.
        __builtin_amdgcn_global_load_async_to_lds_b128(
            (async_gptr)aPtr, (async_lptr)asDst, 0, 0);
        __builtin_amdgcn_global_load_async_to_lds_b128(
            (async_gptr)aPtr, (async_lptr)asDst, 16, 0);
#else
        *(uint4*)(asDst)     = a0;
        *(uint4*)(asDst + 8) = a1;
#endif
        uint4 bq;
        bq.x = (unsigned)bv[0] | ((unsigned)bv[1] << 16);
        bq.y = (unsigned)bv[2] | ((unsigned)bv[3] << 16);
        bq.z = (unsigned)bv[4] | ((unsigned)bv[5] << 16);
        bq.w = (unsigned)bv[6] | ((unsigned)bv[7] << 16);
        *(uint4*)(BsT + bn * 32 + bk8) = bq;
#if HAVE_ASYNC_LDS
        __builtin_amdgcn_s_wait_asynccnt(0);   // A-tile bytes resident in LDS
#endif
        __syncthreads();
        aPtr += 32;
        bPtr += (size_t)32 * N;

        // --- fragments + WMMA ---
        Frag af[2], bf[2];
        #pragma unroll
        for (int i = 0; i < 2; ++i) {
            const int row = wm + i * 16 + l15;
            af[i].q[0] = *(const uint4*)(As + row * 32 + kbA);
            af[i].q[1] = *(const uint4*)(As + row * 32 + kbA + 8);
        }
        #pragma unroll
        for (int j = 0; j < 2; ++j) {
            const int nn = wn + j * 16 + l15;
            bf[j].q[0] = *(const uint4*)(BsT + nn * 32 + ksB);
            bf[j].q[1] = *(const uint4*)(BsT + nn * 32 + ksB + 8);
        }
        #pragma unroll
        for (int i = 0; i < 2; ++i)
            #pragma unroll
            for (int j = 0; j < 2; ++j)
                acc[i][j] = wmma_bf16(af[i].v, bf[j].v, acc[i][j]);
    }

    // --- epilogue: bias + store (row offsets are compile-time immediates) ---
    #pragma unroll
    for (int i = 0; i < 2; ++i)
        #pragma unroll
        for (int j = 0; j < 2; ++j) {
            const int col  = n0 + wn + j * 16 + l15;
            const int row0 = m0 + wm + i * 16 + hi * 8;   // VGPR r -> row0 + r
            const float bb = bias[col];
            if (STORE_BF16) {
                unsigned short* p = (unsigned short*)Cout + (size_t)row0 * N + col;
                #pragma unroll
                for (int r = 0; r < 8; ++r)
                    p[r * N] = f2bf(acc[i][j][r] + bb);
            } else {
                float* p = (float*)Cout + (size_t)row0 * N + col;
                #pragma unroll
                for (int r = 0; r < 8; ++r)
                    p[r * N] = acc[i][j][r] + bb;
            }
        }
}

// ---------------------------------------------------------------------------
// In-place RoPE on bf16 [S, n_head, 64], half-split (rotate_half) convention
// ---------------------------------------------------------------------------
__global__ __launch_bounds__(256)
void rope_kernel(unsigned short* __restrict__ x,
                 const int* __restrict__ positions,
                 int n_head, long long total)
{
    long long idx = (long long)blockIdx.x * blockDim.x + threadIdx.x;
    if (idx >= total) return;
    const int i   = (int)(idx & 31);
    const int h   = (int)((idx >> 5) % n_head);
    const int tok = (int)(idx / (32LL * n_head));
    const float inv_freq = __expf(-((2.0f * (float)i) / 64.0f) * __logf(150000.0f));
    const float ang = (float)positions[tok] * inv_freq;
    float sn, cs;
    __sincosf(ang, &sn, &cs);
    const size_t base = (size_t)tok * ((size_t)n_head * 64) + (size_t)h * 64 + i;
    const float x1 = bf2f(x[base]);
    const float x2 = bf2f(x[base + 32]);
    x[base]      = f2bf(x1 * cs - x2 * sn);
    x[base + 32] = f2bf(x2 * cs + x1 * sn);
}

// ---------------------------------------------------------------------------
// Sliding-window attention, one workgroup per (block b of 128 queries, head h).
// Q,K fragments straight from global (row-major K == B^T layout WMMA wants),
// V transposed into LDS, P re-shaped C->A layout via per-wave LDS scratch.
// ---------------------------------------------------------------------------
__global__ __launch_bounds__(256)
void attn_kernel(const unsigned short* __restrict__ Q,   // [S, 64*64] bf16 (RoPE'd)
                 const unsigned short* __restrict__ Kg,  // [S, 8*64]  bf16 (RoPE'd)
                 const unsigned short* __restrict__ Vg,  // [S, 8*64]  bf16
                 unsigned short* __restrict__ O)         // [S, 64*64] bf16
{
    __shared__ __align__(16) unsigned short VT[64 * 256];   // [hd][key] 32 KB
    __shared__ __align__(16) unsigned short Ps[8][16 * 32]; // per-wave P scratch 8 KB

    const int b    = blockIdx.x;    // query block (64)
    const int h    = blockIdx.y;    // query head (64)
    const int kvh  = h >> 3;        // GQA: 8 q-heads per kv-head
    const int t    = threadIdx.x;
    const int lane = t & 31;
    const int wave = t >> 5;        // 8 waves, each owns 16 query rows
    const int l15  = lane & 15;
    const int hi   = lane >> 4;
    const int kbA  = hi * 16;
    const int ksB  = hi * 16;

    // ---- stage V^T into LDS (keys b*128-128 .. b*128+127; OOB -> 0) ----
    {
        const int hd = t & 63;
        const int kq = t >> 6;      // 0..3, each covers 64 keys
        const unsigned short* vp = Vg + ((size_t)(b * 128 + kq * 64 - 128)) * 512
                                      + (size_t)kvh * 64 + hd;
        for (int kc = 0; kc < 64; kc += 8) {
            unsigned short tmp[8];
            #pragma unroll
            for (int u = 0; u < 8; ++u) {
                const int gk = b * 128 + kq * 64 + kc + u - 128;
                tmp[u] = (gk >= 0) ? vp[(size_t)(kc + u) * 512] : (unsigned short)0;
            }
            uint4 pk;
            pk.x = (unsigned)tmp[0] | ((unsigned)tmp[1] << 16);
            pk.y = (unsigned)tmp[2] | ((unsigned)tmp[3] << 16);
            pk.z = (unsigned)tmp[4] | ((unsigned)tmp[5] << 16);
            pk.w = (unsigned)tmp[6] | ((unsigned)tmp[7] << 16);
            *(uint4*)(VT + hd * 256 + kq * 64 + kc) = pk;
        }
    }
    __syncthreads();

    // ---- Q A-fragments (16 rows x 64 hd = 2 K-steps), direct from global ----
    const int qrow = b * 128 + wave * 16 + l15;
    Frag qa[2];
    #pragma unroll
    for (int kst = 0; kst < 2; ++kst) {
        const unsigned short* p = Q + (size_t)qrow * 4096 + (size_t)h * 64 + kst * 32 + kbA;
        qa[kst].q[0] = *(const uint4*)(p);
        qa[kst].q[1] = *(const uint4*)(p + 8);
    }

    // ---- scores: 16 rows x 256 keys = 16 WMMA C-tiles ----
    const v8f vzero = {0.f,0.f,0.f,0.f,0.f,0.f,0.f,0.f};
    v8f sc[16];
    #pragma unroll
    for (int nt = 0; nt < 16; ++nt) sc[nt] = vzero;

    for (int nt = 0; nt < 16; ++nt) {
        const bool havePrev = (b > 0) || (nt >= 8);   // uniform over workgroup
        if (!havePrev) continue;                      // block 0's prev keys: masked anyway
        const int key = nt * 16 + l15;
        const int gk  = b * 128 + key - 128;
        const unsigned short* kp = Kg + (size_t)gk * 512 + (size_t)kvh * 64;
        #pragma unroll
        for (int kst = 0; kst < 2; ++kst) {
            Frag kf;
            kf.q[0] = *(const uint4*)(kp + kst * 32 + ksB);
            kf.q[1] = *(const uint4*)(kp + kst * 32 + ksB + 8);
            sc[nt] = wmma_bf16(qa[kst].v, kf.v, sc[nt]);
        }
    }

    // ---- mask + softmax (row = wave*16 + j + hi*8, key = nt*16 + l15) ----
    float rsum[8];
    const float scale = 0.125f;   // 1/sqrt(64)
    #pragma unroll
    for (int j = 0; j < 8; ++j) {
        const int rm = wave * 16 + j + hi * 8;   // local query row
        float mx = -3.0e38f;
        #pragma unroll
        for (int nt = 0; nt < 16; ++nt) {
            const int key = nt * 16 + l15;
            const int d = rm - key + 128;        // qpos - kpos
            const bool valid = (d >= 0) && (d < 128) && ((b * 128 + key - 128) >= 0);
            const float v = valid ? sc[nt][j] * scale : -3.0e38f;
            sc[nt][j] = v;
            mx = fmaxf(mx, v);
        }
        #pragma unroll
        for (int off = 8; off >= 1; off >>= 1)
            mx = fmaxf(mx, __shfl_xor(mx, off, 32));   // reduce within 16-lane half
        float sum = 0.f;
        #pragma unroll
        for (int nt = 0; nt < 16; ++nt) {
            const float e = __expf(sc[nt][j] - mx);    // masked -> exp(-huge) = 0
            sc[nt][j] = e;
            sum += e;
        }
        #pragma unroll
        for (int off = 8; off >= 1; off >>= 1)
            sum += __shfl_xor(sum, off, 32);
        rsum[j] = sum;
    }

    // ---- PV: out[16,64] += P[16,256] * V[256,64], K-steps of 32 keys ----
    v8f ov[4];
    #pragma unroll
    for (int jt = 0; jt < 4; ++jt) ov[jt] = vzero;

    for (int kc = 0; kc < 8; ++kc) {
        // write this 16x32 P chunk (C-layout -> row-major LDS)
        #pragma unroll
        for (int half = 0; half < 2; ++half) {
            const int nt = kc * 2 + half;
            #pragma unroll
            for (int j = 0; j < 8; ++j) {
                const int r = j + hi * 8;
                const int c = half * 16 + l15;
                Ps[wave][r * 32 + c] = f2bf(sc[nt][j]);
            }
        }
        // same-wave LDS ops are in-order: store -> load is safe without barrier
        Frag pa;
        pa.q[0] = *(const uint4*)(&Ps[wave][l15 * 32 + kbA]);
        pa.q[1] = *(const uint4*)(&Ps[wave][l15 * 32 + kbA + 8]);
        #pragma unroll
        for (int jt = 0; jt < 4; ++jt) {
            Frag bfv;
            const int hd = jt * 16 + l15;
            bfv.q[0] = *(const uint4*)(VT + hd * 256 + kc * 32 + ksB);
            bfv.q[1] = *(const uint4*)(VT + hd * 256 + kc * 32 + ksB + 8);
            ov[jt] = wmma_bf16(pa.v, bfv.v, ov[jt]);
        }
    }

    // ---- normalize + store bf16 attn output [S, 64*64] ----
    #pragma unroll
    for (int jt = 0; jt < 4; ++jt) {
        unsigned short* op = O + (size_t)(b * 128 + wave * 16 + hi * 8) * 4096
                               + (size_t)h * 64 + jt * 16 + l15;
        #pragma unroll
        for (int j = 0; j < 8; ++j)
            op[j * 4096] = f2bf(ov[jt][j] / rsum[j]);
    }
}

// ---------------------------------------------------------------------------
// Launcher
// ---------------------------------------------------------------------------
extern "C" void kernel_launch(void* const* d_in, const int* in_sizes, int n_in,
                              void* d_out, int out_size, void* d_ws, size_t ws_size,
                              hipStream_t stream)
{
    (void)in_sizes; (void)n_in; (void)out_size; (void)ws_size;

    const float* hidden    = (const float*)d_in[0];
    const int*   positions = (const int*)d_in[1];   // int32 positions
    const float* Wq = (const float*)d_in[2];
    const float* bq = (const float*)d_in[3];
    const float* Wk = (const float*)d_in[4];
    const float* bk = (const float*)d_in[5];
    const float* Wv = (const float*)d_in[6];
    const float* bv = (const float*)d_in[7];
    const float* Wo = (const float*)d_in[8];
    const float* bo = (const float*)d_in[9];

    char* ws = (char*)d_ws;
    unsigned short* h_bf   = (unsigned short*)(ws + OFF_HID);
    unsigned short* wq_bf  = (unsigned short*)(ws + OFF_WQ);
    unsigned short* wk_bf  = (unsigned short*)(ws + OFF_WK);
    unsigned short* wv_bf  = (unsigned short*)(ws + OFF_WV);
    unsigned short* wo_bf  = (unsigned short*)(ws + OFF_WO);
    unsigned short* q_bf   = (unsigned short*)(ws + OFF_Q);
    unsigned short* k_bf   = (unsigned short*)(ws + OFF_K);
    unsigned short* v_bf   = (unsigned short*)(ws + OFF_V);
    unsigned short* att_bf = (unsigned short*)(ws + OFF_ATT);

    // 1) fp32 -> bf16 conversions (vectorized, 8 elems/thread)
    {
        struct Cv { const float* s; unsigned short* d; long long n; };
        const Cv cv[5] = {
            { hidden, h_bf,  (long long)S_TOK * HID_DIM },
            { Wq,     wq_bf, (long long)HID_DIM * 4096  },
            { Wk,     wk_bf, (long long)HID_DIM * 512   },
            { Wv,     wv_bf, (long long)HID_DIM * 512   },
            { Wo,     wo_bf, (long long)4096 * HID_DIM  },
        };
        for (int i = 0; i < 5; ++i) {
            long long n8 = cv[i].n / 8;
            unsigned blocks = (unsigned)((n8 + 255) / 256);
            f32_to_bf16_kernel<<<blocks, 256, 0, stream>>>(cv[i].s, cv[i].d, n8);
        }
    }

    // 2) QKV projections (bf16 out)
    gemm_bf16_kernel<4096, HID_DIM, true><<<dim3(4096 / 64, S_TOK / 128), 256, 0, stream>>>(
        h_bf, wq_bf, bq, q_bf);
    gemm_bf16_kernel<512, HID_DIM, true><<<dim3(512 / 64, S_TOK / 128), 256, 0, stream>>>(
        h_bf, wk_bf, bk, k_bf);
    gemm_bf16_kernel<512, HID_DIM, true><<<dim3(512 / 64, S_TOK / 128), 256, 0, stream>>>(
        h_bf, wv_bf, bv, v_bf);

    // 3) RoPE (in place on bf16 q, k)
    rope_kernel<<<(unsigned)(((long long)S_TOK * NQ_H * 32 + 255) / 256), 256, 0, stream>>>(
        q_bf, positions, NQ_H, (long long)S_TOK * NQ_H * 32);
    rope_kernel<<<(unsigned)(((long long)S_TOK * NKV_H * 32 + 255) / 256), 256, 0, stream>>>(
        k_bf, positions, NKV_H, (long long)S_TOK * NKV_H * 32);

    // 4) sliding-window attention
    attn_kernel<<<dim3(S_TOK / 128, NQ_H), 256, 0, stream>>>(q_bf, k_bf, v_bf, att_bf);

    // 5) output projection (fp32 out to d_out)
    gemm_bf16_kernel<HID_DIM, 4096, false><<<dim3(HID_DIM / 64, S_TOK / 128), 256, 0, stream>>>(
        att_bf, wo_bf, bo, (float*)d_out);
}
